// VanillaRNNScratch_69655779606696
// MI455X (gfx1250) — matrix operand order repaired and checked by reading, not compiled, not run on hardware
//
#include <hip/hip_runtime.h>
#include <hip/hip_bf16.h>
#include <stdint.h>

// ---------------------------------------------------------------------------
// VanillaRNN for MI455X (gfx1250), wave32, WMMA bf16 (16x16x32, f32 accum).
//
//   B=64, S=512, H=1024, V=128
//   Phase A: convert/transpose weights to bf16, init barrier state.
//   Phase B: embW[v][j] = sum_k emb[v][k]*W_hx[k][j] + b_hx[j]   (f32, WMMA)
//   Phase C: persistent recurrent kernel, 64 WGs x 256 thr, W_hh slice in LDS
//            (loaded once by the Tensor Data Mover, reused 512 steps),
//            grid barrier per timestep.
//   Phase D: logits = hs @ W_out + b_out (WMMA).
// ---------------------------------------------------------------------------

typedef __attribute__((ext_vector_type(16))) __bf16 v16bf;
typedef __attribute__((ext_vector_type(8)))  float  v8f;
typedef __attribute__((ext_vector_type(4)))  unsigned int v4u;
typedef __attribute__((ext_vector_type(8)))  int    v8i_t;
typedef __attribute__((ext_vector_type(4)))  int    v4i_t;

#define RNN_B 64
#define RNN_S 512
#define RNN_H 1024
#define RNN_V 128
#define RNN_BH (RNN_B * RNN_H)

// workspace layout (bytes)
#define OFF_EMB_BF  ((size_t)0)                         // 128*1024*2
#define OFF_WHX_T   (OFF_EMB_BF + 262144)               // 1024*1024*2
#define OFF_WHH_T   (OFF_WHX_T  + 2097152)              // 1024*1024*2
#define OFF_WOUT_T  (OFF_WHH_T  + 2097152)              // 128*1024*2
#define OFF_H0B     (OFF_WOUT_T + 262144)               // 64*1024*2
#define OFF_EMBW    (OFF_H0B    + 131072)               // 128*1024*4
#define OFF_SYNC    (OFF_EMBW   + 524288)               // barrier cnt/flag
#define OFF_HS      (OFF_SYNC   + 256)                  // 512*64*1024*2

union AFrag { uint4 q[2]; v16bf v; };

// A fragment (16x32 bf16, row-major source, ld = K stride in elements).
// Lanes 0-15: row M=lane, K = k0+[0..8) and k0+[16..24)
// Lanes 16-31: row M=lane-16, K = k0+[8..16) and k0+[24..32)
__device__ __forceinline__ v16bf load_a_frag(const __bf16* base, int ld,
                                             int k0, int lane) {
    int row = lane & 15, half = lane >> 4;
    const __bf16* p = base + (size_t)row * ld + k0 + half * 8;
    AFrag f;
    f.q[0] = *(const uint4*)p;
    f.q[1] = *(const uint4*)(p + 16);
    return f.v;
}

// B fragment (32x16 bf16) from a pre-transposed matrix BT[n][k] (ld = K).
// Lanes 0-15 hold K=k0+[0..8),[16..24) of column n=lane; lanes 16-31 hold K=k0+[8..16),[24..32).
__device__ __forceinline__ v16bf load_bt_frag(const __bf16* bt, int ld,
                                              int k0, int lane) {
    int col = lane & 15, half = lane >> 4;
    const __bf16* p = bt + (size_t)col * ld + k0 + half * 8;
    AFrag f;
    f.q[0] = *(const uint4*)p;
    f.q[1] = *(const uint4*)(p + 16);
    return f.v;
}

__device__ __forceinline__ v8f wmma_bf16(v16bf a, v16bf b, v8f c) {
    return __builtin_amdgcn_wmma_f32_16x16x32_bf16(false, a, false, b,
                                                   (short)0, c, false, false);
}

// ---------------------------------------------------------------------------
// Phase A: convert + transpose weights to bf16, init grid-barrier state.
// ---------------------------------------------------------------------------
__global__ void rnn_prep_kernel(const float* __restrict__ h0,
                                const float* __restrict__ emb,
                                const float* __restrict__ W_hx,
                                const float* __restrict__ W_hh,
                                const float* __restrict__ W_out,
                                char* __restrict__ ws) {
    __bf16* emb_b  = (__bf16*)(ws + OFF_EMB_BF);
    __bf16* whx_t  = (__bf16*)(ws + OFF_WHX_T);
    __bf16* whh_t  = (__bf16*)(ws + OFF_WHH_T);
    __bf16* wout_t = (__bf16*)(ws + OFF_WOUT_T);
    __bf16* h0b    = (__bf16*)(ws + OFF_H0B);
    unsigned* sync = (unsigned*)(ws + OFF_SYNC);

    int i = blockIdx.x * blockDim.x + threadIdx.x;   // exactly 1024*1024 threads
    {   // W_hx, W_hh : [K=1024][N=1024] -> T[N][K] bf16
        int k = i >> 10, n = i & 1023;
        whx_t[(size_t)n * RNN_H + k] = (__bf16)W_hx[i];
        whh_t[(size_t)n * RNN_H + k] = (__bf16)W_hh[i];
    }
    if (i < RNN_V * RNN_H) {
        emb_b[i] = (__bf16)emb[i];
        int k = i >> 7, v = i & 127;                 // W_out [1024][128] -> T[128][1024]
        wout_t[(size_t)v * RNN_H + k] = (__bf16)W_out[i];
    }
    if (i < RNN_B * RNN_H) h0b[i] = (__bf16)h0[i];
    if (i == 0) { sync[0] = 0u; sync[1] = 0u; }
}

// ---------------------------------------------------------------------------
// Phase B: embW = emb @ W_hx + b_hx  (128 x 1024, f32).  512 wave-jobs.
// ---------------------------------------------------------------------------
__global__ void rnn_embw_kernel(const float* __restrict__ b_hx,
                                char* __restrict__ ws) {
    const __bf16* emb_b = (const __bf16*)(ws + OFF_EMB_BF);
    const __bf16* whx_t = (const __bf16*)(ws + OFF_WHX_T);
    float* embW = (float*)(ws + OFF_EMBW);

    int lane = threadIdx.x & 31;
    int wid  = blockIdx.x * (blockDim.x >> 5) + (threadIdx.x >> 5); // 0..511
    int mt = wid >> 6;        // 0..7   (M = 128)
    int nt = wid & 63;        // 0..63  (N = 1024)

    const __bf16* abase = emb_b + (size_t)(mt * 16) * RNN_H;
    const __bf16* bbase = whx_t + (size_t)(nt * 16) * RNN_H;
    v8f c = {};
    for (int k0 = 0; k0 < RNN_H; k0 += 32) {
        v16bf a = load_a_frag(abase, RNN_H, k0, lane);
        v16bf b = load_bt_frag(bbase, RNN_H, k0, lane);
        c = wmma_bf16(a, b, c);
    }
    int col = nt * 16 + (lane & 15);
    float bias = b_hx[col];
    #pragma unroll
    for (int r = 0; r < 8; ++r) {
        int row = mt * 16 + r + ((lane >> 4) ? 8 : 0);
        embW[(size_t)row * RNN_H + col] = c[r] + bias;
    }
}

// ---------------------------------------------------------------------------
// Phase C: persistent recurrent kernel. 64 blocks x 256 threads.
// Block wg owns hidden columns [wg*16, wg*16+16). W_hh_T slice lives in LDS.
// Waves 0-3: M-tile m, K in [0,512). Waves 4-7: same M-tile, K in [512,1024).
// ---------------------------------------------------------------------------
__device__ __forceinline__ void grid_barrier(unsigned* cnt, unsigned* flag,
                                             unsigned nblocks, unsigned gen) {
    __syncthreads();
    if (threadIdx.x == 0) {
        unsigned prev = __hip_atomic_fetch_add(cnt, 1u, __ATOMIC_ACQ_REL,
                                               __HIP_MEMORY_SCOPE_AGENT);
        if (prev == nblocks - 1u) {
            __hip_atomic_store(cnt, 0u, __ATOMIC_RELAXED, __HIP_MEMORY_SCOPE_AGENT);
            __hip_atomic_store(flag, gen, __ATOMIC_RELEASE, __HIP_MEMORY_SCOPE_AGENT);
        } else {
            while (__hip_atomic_load(flag, __ATOMIC_ACQUIRE,
                                     __HIP_MEMORY_SCOPE_AGENT) < gen) { }
        }
    }
    __syncthreads();
}

__global__ void __launch_bounds__(256)
rnn_recurrent_kernel(const int* __restrict__ x,       // [B][S]
                     char* __restrict__ ws,
                     float* __restrict__ h_final) {   // d_out + B*S*V
    __shared__ __align__(16) uint4 sB4[16 * 128];     // 16 cols x 1024 K bf16 = 32KB
    __shared__ float sRed[4 * 256];                   // split-K partials

    const __bf16* whh_t = (const __bf16*)(ws + OFF_WHH_T);
    const __bf16* h0b   = (const __bf16*)(ws + OFF_H0B);
    const float*  embW  = (const float*)(ws + OFF_EMBW);
    __bf16* hs          = (__bf16*)(ws + OFF_HS);
    unsigned* sync      = (unsigned*)(ws + OFF_SYNC);

    const int wg   = blockIdx.x;          // 0..63 -> columns n0..n0+15
    const int n0   = wg * 16;
    const int tid  = threadIdx.x;
    const int lane = tid & 31;
    const int wv   = tid >> 5;            // 0..7
    const int m    = wv & 3;              // M-tile (batch rows m*16..m*16+15)
    const int kh   = wv >> 2;             // K half: 0 -> [0,512), 1 -> [512,1024)

    // ---- Load resident W_hh^T slice (32KB contiguous) into LDS ----
    const __bf16* gslice = whh_t + (size_t)n0 * RNN_H;
#if __has_builtin(__builtin_amdgcn_tensor_load_to_lds) && __has_builtin(__builtin_amdgcn_s_wait_tensorcnt)
    if (wv == 0) {
        uint32_t ldsaddr = (uint32_t)(size_t)(&sB4[0]);
        uint64_t ga = (uint64_t)(size_t)gslice;
        v4u g0 = { 1u,                                   // count=1, user descriptor
                   ldsaddr,                              // lds_addr
                   (uint32_t)ga,                         // global_addr lo
                   ((uint32_t)(ga >> 32) & 0x01FFFFFFu) | 0x80000000u }; // type=2
        // 2D tensor: 16384 x 1 elements of 2B, tile 16384 x 1, stride 16384
        v8i_t g1 = { (int)0x00010000,   // data_size=1 (2B), mask=0
                     (int)0x40000000,   // tensor_dim0 lo16 = 16384
                     (int)0x00010000,   // tensor_dim0 hi=0 | tensor_dim1 lo16 = 1
                     (int)0x40000000,   // tensor_dim1 hi=0 | tile_dim0 = 16384
                     (int)0x00000001,   // tile_dim1 = 1, tile_dim2 = 0
                     (int)16384,        // tensor_dim0_stride lo32
                     (int)0x40000000,   // stride0 hi=0 | tensor_dim1_stride lo16
                     (int)0 };
        v4i_t gz4 = { 0, 0, 0, 0 };
        v8i_t gz8 = { 0, 0, 0, 0, 0, 0, 0, 0 };
        __builtin_amdgcn_tensor_load_to_lds(g0, g1, gz4, gz4, gz8, 0);
        __builtin_amdgcn_s_wait_tensorcnt(0);
    }
#else
    for (int i = tid; i < 16 * 128; i += 256)
        sB4[i] = ((const uint4*)gslice)[i];
#endif
    __syncthreads();

    unsigned* cnt  = sync;
    unsigned* flag = sync + 1;

    for (int t = 0; t < RNN_S; ++t) {
        const __bf16* prev = (t == 0) ? h0b : (hs + (size_t)(t - 1) * RNN_BH);
        const __bf16* abase = prev + (size_t)(m * 16) * RNN_H;

        v8f c = {};
        const int kbase = kh * 512;
        #pragma unroll 4
        for (int ks = 0; ks < 16; ++ks) {
            int k0 = kbase + ks * 32;
            v16bf a = load_a_frag(abase, RNN_H, k0, lane);
            // B fragment straight from LDS (ds_load_b128 x2)
            int col  = lane & 15, halfk = lane >> 4;
            int bidx = col * 128 + (k0 >> 3) + halfk;
            AFrag bf;
            bf.q[0] = sB4[bidx];
            bf.q[1] = sB4[bidx + 2];
            c = wmma_bf16(a, bf.v, c);
        }

        if (kh == 1) {      // publish upper-K partial
            #pragma unroll
            for (int r = 0; r < 8; ++r) sRed[m * 256 + r * 32 + lane] = c[r];
        }
        __syncthreads();
        if (kh == 0) {      // reduce, bias-gather, tanh, store
            int col = lane & 15;
            #pragma unroll
            for (int r = 0; r < 8; ++r) {
                float acc = c[r] + sRed[m * 256 + r * 32 + lane];
                int brow = m * 16 + r + ((lane >> 4) ? 8 : 0);
                int xb = x[brow * RNN_S + t];
                float e = embW[(size_t)xb * RNN_H + n0 + col];
                float hv = tanhf(acc + e);
                hs[(size_t)t * RNN_BH + (size_t)brow * RNN_H + n0 + col] = (__bf16)hv;
                if (t == RNN_S - 1)
                    h_final[(size_t)brow * RNN_H + n0 + col] = hv;
            }
        }
        grid_barrier(cnt, flag, gridDim.x, (unsigned)(t + 1));
    }
}

// ---------------------------------------------------------------------------
// Phase D: logits[b][t][:] = hs[t][b][:] @ W_out + b_out.
// 2048 blocks (one 16-row M-tile each) x 8 waves (one 16-col N-tile each).
// ---------------------------------------------------------------------------
__global__ void __launch_bounds__(256)
rnn_logits_kernel(const float* __restrict__ b_out,
                  char* __restrict__ ws,
                  float* __restrict__ logits) {
    const __bf16* hs     = (const __bf16*)(ws + OFF_HS);
    const __bf16* wout_t = (const __bf16*)(ws + OFF_WOUT_T);

    int lane = threadIdx.x & 31;
    int wv   = threadIdx.x >> 5;          // N-tile 0..7
    int mt   = blockIdx.x;                // 0..2047

    const __bf16* abase = hs + (size_t)(mt * 16) * RNN_H;
    const __bf16* bbase = wout_t + (size_t)(wv * 16) * RNN_H;

    v8f c = {};
    #pragma unroll 4
    for (int k0 = 0; k0 < RNN_H; k0 += 32) {
        v16bf a = load_a_frag(abase, RNN_H, k0, lane);
        v16bf b = load_bt_frag(bbase, RNN_H, k0, lane);
        c = wmma_bf16(a, b, c);
    }
    int col = wv * 16 + (lane & 15);
    float bias = b_out[col];
    #pragma unroll
    for (int r = 0; r < 8; ++r) {
        int grow = mt * 16 + r + ((lane >> 4) ? 8 : 0);  // = t*64 + b
        int t = grow >> 6, b = grow & 63;
        logits[((size_t)b * RNN_S + t) * RNN_V + col] = c[r] + bias;
    }
}

// ---------------------------------------------------------------------------
extern "C" void kernel_launch(void* const* d_in, const int* in_sizes, int n_in,
                              void* d_out, int out_size, void* d_ws, size_t ws_size,
                              hipStream_t stream) {
    const int*   x     = (const int*)d_in[0];
    const float* h0    = (const float*)d_in[1];
    const float* emb   = (const float*)d_in[2];
    const float* W_hx  = (const float*)d_in[3];
    const float* b_hx  = (const float*)d_in[4];
    const float* W_hh  = (const float*)d_in[5];
    const float* W_out = (const float*)d_in[6];
    const float* b_out = (const float*)d_in[7];

    float* logits  = (float*)d_out;                       // [B][S][V]
    float* h_final = logits + (size_t)RNN_B * RNN_S * RNN_V;
    char*  ws      = (char*)d_ws;

    rnn_prep_kernel<<<4096, 256, 0, stream>>>(h0, emb, W_hx, W_hh, W_out, ws);
    rnn_embw_kernel<<<64, 256, 0, stream>>>(b_hx, ws);
    rnn_recurrent_kernel<<<64, 256, 0, stream>>>(x, ws, h_final);
    rnn_logits_kernel<<<2048, 256, 0, stream>>>(b_out, ws, logits);
}
